// GraphAttentionalLayer_37572373905987
// MI455X (gfx1250) — matrix operand order repaired and checked by reading, not compiled, not run on hardware
//
#include <hip/hip_runtime.h>
#include <hip/hip_bf16.h>

// ---------- types ----------
typedef __attribute__((ext_vector_type(16))) __bf16  v16bf;
typedef __attribute__((ext_vector_type(8)))  __bf16  bf16x8;
typedef __attribute__((ext_vector_type(8)))  float   v8f;
typedef __attribute__((ext_vector_type(4)))  float   f4;

#define NN   6144
#define INF_ 512
#define OUTF 256
#define HEADS 4
#define NEG_INF_F (-9.0e15f)

// ---------- helpers ----------
static __device__ __forceinline__ v16bf combine8(bf16x8 lo, bf16x8 hi) {
  v16bf r;
#pragma unroll
  for (int i = 0; i < 8; ++i) { r[i] = lo[i]; r[i + 8] = hi[i]; }
  return r;
}
static __device__ __forceinline__ v8f wmma_bf16(v16bf a, v16bf b, v8f c) {
  return __builtin_amdgcn_wmma_f32_16x16x32_bf16(
      false, a, false, b, (short)0, c, false, false);
}
static __device__ __forceinline__ float leaky(float x) {
  return x > 0.0f ? x : 0.1f * x;
}

// ---------- k0: W (512x1024 f32) -> W_T (1024x512 bf16) ----------
__global__ __launch_bounds__(256) void k0_transW(const float* __restrict__ W,
                                                 __bf16* __restrict__ wT) {
  int idx = blockIdx.x * 256 + threadIdx.x;          // < 512*1024
  int k = idx >> 10, c = idx & 1023;
  wT[c * INF_ + k] = (__bf16)W[idx];
}

// ---------- k1: HT_T[c][n] = (h @ W)^T  via bf16 WMMA ----------
__global__ __launch_bounds__(256) void k1_gemm(const float* __restrict__ h,
                                               const __bf16* __restrict__ wT,
                                               __bf16* __restrict__ htT) {
  const int tid = threadIdx.x, lane = tid & 31, w = tid >> 5;
  const int tileId = blockIdx.x * 8 + w;             // < 384*64
  const int tm = tileId >> 6, tc = tileId & 63;
  const int rowA = tm * 16 + (lane & 15);
  const int colB = tc * 16 + (lane & 15);
  const int half = lane >> 4;
  v8f acc = {};
#pragma unroll 4
  for (int kb = 0; kb < INF_; kb += 32) {
    // A fragment: row rowA, K = kb + half*16 + {0..15}
    const f4* ap = (const f4*)(h + (size_t)rowA * INF_ + kb + half * 16);
    f4 a0 = ap[0], a1 = ap[1];
    const f4* ap2 = (const f4*)(h + (size_t)rowA * INF_ + kb + half * 16 + 8);
    f4 a2 = ap2[0], a3 = ap2[1];
    v16bf A;
#pragma unroll
    for (int i = 0; i < 4; ++i) {
      A[i]      = (__bf16)a0[i];
      A[i + 4]  = (__bf16)a1[i];
      A[i + 8]  = (__bf16)a2[i];
      A[i + 12] = (__bf16)a3[i];
    }
    // B fragment: col colB, K = kb + half*16 + {0..15}, contiguous in W_T
    const bf16x8* bp = (const bf16x8*)(wT + (size_t)colB * INF_ + kb + half * 16);
    v16bf B = combine8(bp[0], bp[1]);
    acc = wmma_bf16(A, B, acc);
  }
  // store transposed: HT_T[c][node]
#pragma unroll
  for (int r = 0; r < 8; ++r) {
    int node = tm * 16 + r + half * 8;
    int c    = tc * 16 + (lane & 15);
    htT[(size_t)c * NN + node] = (__bf16)acc[r];
  }
}

// ---------- k1b: src/tgt projections ----------
__global__ __launch_bounds__(256) void k1b_srctgt(const __bf16* __restrict__ htT,
                                                  const float* __restrict__ a,
                                                  float* __restrict__ src,
                                                  float* __restrict__ tgt) {
  int gid = blockIdx.x * 256 + threadIdx.x;          // < HEADS*NN
  int hh = gid / NN, n = gid % NN;
  float s = 0.0f, t = 0.0f;
#pragma unroll 4
  for (int d = 0; d < OUTF; ++d) {
    float v = (float)htT[(size_t)(hh * OUTF + d) * NN + n];
    s += v * a[hh * (2 * OUTF) + d];
    t += v * a[hh * (2 * OUTF) + OUTF + d];
  }
  src[hh * NN + n] = s;
  tgt[hh * NN + n] = t;
}

// ---------- k2: per-row online softmax stats (all heads, one adj pass) ----------
__global__ __launch_bounds__(256) void k2_rowstats(const int* __restrict__ adj,
                                                   const float* __restrict__ src,
                                                   const float* __restrict__ tgt,
                                                   float* __restrict__ mrow,
                                                   float* __restrict__ srow) {
  const int wid  = (blockIdx.x * 256 + threadIdx.x) >> 5;  // row, < NN
  const int lane = threadIdx.x & 31;
  float sc[HEADS], m[HEADS], s[HEADS];
#pragma unroll
  for (int hh = 0; hh < HEADS; ++hh) {
    sc[hh] = src[hh * NN + wid];
    m[hh] = -3.0e38f;
    s[hh] = 0.0f;
  }
  const int* arow = adj + (size_t)wid * NN;
  for (int j = lane; j < NN; j += 32) {
    int av = arow[j];
    __builtin_prefetch(arow + j + 256, 0, 1);
#pragma unroll
    for (int hh = 0; hh < HEADS; ++hh) {
      float e = (av > 0) ? leaky(sc[hh] + tgt[hh * NN + j]) : NEG_INF_F;
      if (e > m[hh]) { s[hh] = s[hh] * __expf(m[hh] - e) + 1.0f; m[hh] = e; }
      else           { s[hh] += __expf(e - m[hh]); }
    }
  }
#pragma unroll
  for (int off = 16; off > 0; off >>= 1) {
#pragma unroll
    for (int hh = 0; hh < HEADS; ++hh) {
      float mo = __shfl_xor(m[hh], off, 32);
      float so = __shfl_xor(s[hh], off, 32);
      float nm = fmaxf(m[hh], mo);
      s[hh] = s[hh] * __expf(m[hh] - nm) + so * __expf(mo - nm);
      m[hh] = nm;
    }
  }
  if (lane == 0) {
#pragma unroll
    for (int hh = 0; hh < HEADS; ++hh) {
      mrow[hh * NN + wid] = m[hh];
      srow[hh * NN + wid] = s[hh];
    }
  }
}

// ---------- k3: fused masked-softmax @ ht, mean over heads (WMMA) ----------
__global__ __launch_bounds__(256) void k3_attn(const int* __restrict__ adj,
                                               const float* __restrict__ src,
                                               const float* __restrict__ mrow,
                                               const float* __restrict__ srow,
                                               const float* __restrict__ tgt,
                                               const __bf16* __restrict__ htT,
                                               float* __restrict__ out) {
  __shared__ __align__(16) __bf16 pS[2][HEADS][16][32];       // double-buffered probs
  __shared__ float srcS[HEADS][16], mS[HEADS][16], scS[HEADS][16];

  const int i0 = blockIdx.x * 16;
  const int tid = threadIdx.x, lane = tid & 31, w = tid >> 5;
  const int half = lane >> 4;
  const int d0 = w * 32;                 // this wave's 32 output columns

  if (tid < HEADS * 16) {
    int hh = tid >> 4, r = tid & 15;
    srcS[hh][r] = src[hh * NN + i0 + r];
    mS[hh][r]   = mrow[hh * NN + i0 + r];
    scS[hh][r]  = 0.25f / srow[hh * NN + i0 + r];   // fold mean over 4 heads
  }
  __syncthreads();

  // p-tile cooperative indices: 4*16*32 = 2048 = 256 threads * 8 values
  const int ph   = tid >> 6;
  const int prow = (tid >> 2) & 15;
  const int pk0  = (tid & 3) * 8;
  const int*   aBase = adj + (size_t)(i0 + prow) * NN + pk0;
  const float* tBase = tgt + ph * NN + pk0;
  const float  sv  = srcS[ph][prow];
  const float  mv  = mS[ph][prow];
  const float  scv = scS[ph][prow];

  auto compute_p = [&](int jcc, __bf16 (*buf)[16][32]) {
    const int4* ap = (const int4*)(aBase + jcc);
    int4 A0 = ap[0], A1 = ap[1];
    __builtin_prefetch(aBase + jcc + 64, 0, 1);       // global_prefetch next adj chunk
    const f4* tp = (const f4*)(tBase + jcc);
    f4 t0 = tp[0], t1 = tp[1];
    int   av[8] = {A0.x, A0.y, A0.z, A0.w, A1.x, A1.y, A1.z, A1.w};
    float tv[8] = {t0[0], t0[1], t0[2], t0[3], t1[0], t1[1], t1[2], t1[3]};
#pragma unroll
    for (int t = 0; t < 8; ++t) {
      float x = sv + tv[t];
      float e = (av[t] > 0) ? leaky(x) : NEG_INF_F;
      buf[ph][prow][pk0 + t] = (__bf16)(__expf(e - mv) * scv);
    }
  };

  v8f acc0 = {}, acc1 = {};
  compute_p(0, pS[0]);
  int cur = 0;
  for (int jc = 0; jc < NN; jc += 32) {
    __syncthreads();                                  // publish pS[cur], retire reads of pS[cur^1]
    if (jc + 32 < NN) compute_p(jc + 32, pS[cur ^ 1]);
#pragma unroll
    for (int hh = 0; hh < HEADS; ++hh) {
      // A fragment from LDS (aligned b128 pairs)
      bf16x8 alo = *(const bf16x8*)&pS[cur][hh][lane & 15][half * 16];
      bf16x8 ahi = *(const bf16x8*)&pS[cur][hh][lane & 15][half * 16 + 8];
      v16bf A = combine8(alo, ahi);
      // B fragments: columns of ht == contiguous runs in HT_T (L2-resident)
      const int col = hh * OUTF + d0 + (lane & 15);
      const int jb  = jc + half * 16;
      const bf16x8* b0 = (const bf16x8*)(htT + (size_t)col * NN + jb);
      const bf16x8* b1 = (const bf16x8*)(htT + (size_t)(col + 16) * NN + jb);
      v16bf B0 = combine8(b0[0], b0[1]);
      v16bf B1 = combine8(b1[0], b1[1]);
      acc0 = wmma_bf16(A, B0, acc0);
      acc1 = wmma_bf16(A, B1, acc1);
    }
    cur ^= 1;
  }
#pragma unroll
  for (int r = 0; r < 8; ++r) {
    int row = i0 + r + half * 8;
    int col = d0 + (lane & 15);
    out[(size_t)row * OUTF + col]      = acc0[r];
    out[(size_t)row * OUTF + col + 16] = acc1[r];
  }
}

// ---------- launch ----------
extern "C" void kernel_launch(void* const* d_in, const int* in_sizes, int n_in,
                              void* d_out, int out_size, void* d_ws, size_t ws_size,
                              hipStream_t stream) {
  const float* h   = (const float*)d_in[0];   // (6144, 512)
  const int*   adj = (const int*)d_in[1];     // (6144, 6144)
  const float* W   = (const float*)d_in[2];   // (512, 1024)
  const float* a   = (const float*)d_in[3];   // (4, 512, 1)
  float* out = (float*)d_out;                 // (6144, 256)

  char* ws = (char*)d_ws;
  __bf16* wT  = (__bf16*)(ws);                         // 1,048,576 B
  __bf16* htT = (__bf16*)(ws + 1048576);               // 12,582,912 B
  float* src  = (float*)(ws + 13631488);               // 98,304 B
  float* tgt  = (float*)(ws + 13729792);               // 98,304 B
  float* mrow = (float*)(ws + 13828096);               // 98,304 B
  float* srow = (float*)(ws + 13926400);               // 98,304 B

  k0_transW  <<<2048, 256, 0, stream>>>(W, wT);
  k1_gemm    <<<3072, 256, 0, stream>>>(h, wT, htT);
  k1b_srctgt <<<96,   256, 0, stream>>>(htT, a, src, tgt);
  k2_rowstats<<<768,  256, 0, stream>>>(adj, src, tgt, mrow, srow);
  k3_attn    <<<384,  256, 0, stream>>>(adj, src, mrow, srow, tgt, htT, out);
}
